// TitanLLaMAAttention_704374636634
// MI455X (gfx1250) — compile-verified, matrix operand order checked by reading, not run on hardware
//
#include <hip/hip_runtime.h>
#include <hip/hip_bf16.h>

// ---------------------------------------------------------------------------
// TitanLLaMA segmented attention forward for gfx1250 (MI455X), wave32 + WMMA.
// GEMMs: v_wmma_f32_16x16x32_f16 with B panels staged through LDS using
// CDNA5 async global->LDS copies (ASYNCcnt) in a double buffer.
// ---------------------------------------------------------------------------

typedef __attribute__((ext_vector_type(16))) _Float16 v16h;
typedef __attribute__((ext_vector_type(8)))  _Float16 v8h;
typedef __attribute__((ext_vector_type(8)))  float    v8f;

#define NN   4128     // sequence length
#define DD   2048     // model dim
#define HH   32       // heads
#define DHD  64       // dim per head
#define PPM  4        // persistent mem tokens
#define SLW  516      // total segment len
#define WWN  8        // windows
#define M3   6144     // 3*H*DH
#define NKEY 1036     // P + 2*SL

// K index held by (lane, element t) of a 16-bit WMMA A/B fragment (ISA 7.12.2)
__device__ __forceinline__ int wmma_k_of_t(int lane, int t) {
    int base = (t < 8) ? t : (t + 8);          // 0..7 then 16..23
    return base + ((lane >> 4) << 3);          // +8 for upper half-wave
}

// Load one half-row-pair fragment: elements t0..7 at p[0..7], t8..15 at p[16..23]
__device__ __forceinline__ v16h frag16(const _Float16* p) {
    v8h lo = *(const v8h*)(p);
    v8h hi = *(const v8h*)(p + 16);
    return __builtin_shufflevector(lo, hi, 0,1,2,3,4,5,6,7,8,9,10,11,12,13,14,15);
}

// ---------------------------------------------------------------------------
// RMSNorm: x = hs * rsqrt(mean(hs^2)+eps) * w   -> f16
// ---------------------------------------------------------------------------
__global__ void __launch_bounds__(256)
rmsnorm_kernel(const float* __restrict__ hs, const float* __restrict__ w,
               _Float16* __restrict__ xh) {
    const int n = blockIdx.x;
    const float* row = hs + (size_t)n * DD;
    float ss = 0.f;
    for (int d = threadIdx.x; d < DD; d += 256) { float v = row[d]; ss += v * v; }
    for (int m = 16; m; m >>= 1) ss += __shfl_xor(ss, m, 32);
    __shared__ float red[8];
    __shared__ float stot;
    if ((threadIdx.x & 31) == 0) red[threadIdx.x >> 5] = ss;
    __syncthreads();
    if (threadIdx.x == 0) {
        float t = 0.f;
        for (int i = 0; i < 8; ++i) t += red[i];
        stot = rsqrtf(t / (float)DD + 1e-6f);
    }
    __syncthreads();
    const float sc = stot;
    for (int d = threadIdx.x; d < DD; d += 256)
        xh[(size_t)n * DD + d] = (_Float16)(row[d] * sc * w[d]);
}

// ---------------------------------------------------------------------------
// Pack f32 weight [K x M] row-major into WMMA B-fragment order (f16):
// Bp[(((panel*(K/32))+kc)*32 + lane)*16 + t] = W[kc*32 + kmap(lane,t)][panel*16 + lane&15]
// ---------------------------------------------------------------------------
__global__ void __launch_bounds__(256)
pack_b_kernel(const float* __restrict__ Wsrc, _Float16* __restrict__ Bp,
              int K, int Mcols) {
    size_t idx = (size_t)blockIdx.x * 256 + threadIdx.x;
    size_t total = (size_t)K * Mcols;
    if (idx >= total) return;
    int t    = (int)(idx & 15);
    int lane = (int)((idx >> 4) & 31);
    size_t rest = idx >> 9;
    int kch = K >> 5;
    int kc    = (int)(rest % kch);
    int panel = (int)(rest / kch);
    int c = panel * 16 + (lane & 15);
    int k = kc * 32 + wmma_k_of_t(lane, t);
    Bp[idx] = (_Float16)Wsrc[(size_t)k * Mcols + c];
}

// Persistent memory k/v f32 -> f16
__global__ void __launch_bounds__(256)
pm_kernel(const float* __restrict__ pmem, _Float16* __restrict__ pk,
          _Float16* __restrict__ pv) {
    int i = blockIdx.x * 256 + threadIdx.x;
    const int sz = HH * PPM * DHD;
    if (i < sz) {
        pk[i] = (_Float16)pmem[i];
        pv[i] = (_Float16)pmem[sz + i];
    }
}

// ---------------------------------------------------------------------------
// WMMA GEMM: C[Mrows x Ncols](f32) = A[Mrows x K](f16 row-major) * Bp(packed)
// 1 wave -> 16x64 tile; 4 waves/block -> 64x64.
// The shared 64-col B panel for each 32-deep k-chunk (4 KB) is staged into an
// LDS double buffer with CDNA5 async global->LDS copies; the copy of chunk
// kc+1 overlaps the WMMAs of chunk kc. ASYNCcnt tracks completion.
// ---------------------------------------------------------------------------
__global__ void __launch_bounds__(128)
wmma_gemm_kernel(const _Float16* __restrict__ A, const _Float16* __restrict__ Bp,
                 float* __restrict__ C, int Mrows, int Ncols, int K) {
    __shared__ __align__(32) _Float16 sB[2][2048];   // 2 x 4 KB double buffer
    const int lane = threadIdx.x & 31;
    const int wave = threadIdx.x >> 5;
    int rowTile = blockIdx.x * 4 + wave;
    const bool active = (rowTile * 16 < Mrows);
    if (!active) rowTile = (Mrows >> 4) - 1;     // clamp: keep wave for barriers/copies
    const int row0 = rowTile * 16;
    const int col0 = blockIdx.y * 64;
    const int r = row0 + (lane & 15);
    const int khalf = (lane >> 4) << 3;
    const int kch = K >> 5;
    const _Float16* arow = A + (size_t)r * K;

    // This thread owns the 32B fragment (panel = wave, lane) of each k-chunk.
    const _Float16* gbase =
        Bp + ((size_t)((col0 >> 4) + wave) * kch) * 512 + (size_t)lane * 16;
    // ISA 10.2: low 32 bits of a flat LDS address are the LDS byte offset.
    const unsigned ldsoff =
        (unsigned)(size_t)(&sB[0][0]) + (unsigned)((wave * 32 + lane) * 32);

    auto issue_copy = [&](int kc, int buf) {
        const _Float16* g = gbase + (size_t)kc * 512;
        unsigned l0 = ldsoff + (unsigned)buf * 4096u;
        // INST_OFFSET is added to BOTH the LDS and global addresses (ISA 10.7)
        asm volatile("global_load_async_to_lds_b128 %0, %1, off"
                     :: "v"(l0), "v"(g) : "memory");
        asm volatile("global_load_async_to_lds_b128 %0, %1, off offset:16"
                     :: "v"(l0), "v"(g) : "memory");
    };

    issue_copy(0, 0);

    v8f acc[4] = {};
    for (int kc = 0; kc < kch; ++kc) {
        __syncthreads();                  // all waves done reading buf[(kc+1)&1]
        const bool more = (kc + 1 < kch);
        if (more) {
            issue_copy(kc + 1, (kc + 1) & 1);
            asm volatile("s_wait_asynccnt 0x2" ::: "memory");  // chunk kc landed
        } else {
            asm volatile("s_wait_asynccnt 0x0" ::: "memory");
        }
        __syncthreads();                  // every thread's chunk-kc copy visible

        v16h af = frag16(arow + kc * 32 + khalf);
        const _Float16* bbuf = &sB[kc & 1][0];
#pragma unroll
        for (int c = 0; c < 4; ++c) {
            v16h bf = *(const v16h*)(bbuf + (c * 32 + lane) * 16);
            acc[c] = __builtin_amdgcn_wmma_f32_16x16x32_f16(
                false, af, false, bf, (short)0, acc[c], false, false);
        }
    }

    if (!active) return;
    const int half = lane >> 4;
    const int cn = lane & 15;
#pragma unroll
    for (int c = 0; c < 4; ++c)
#pragma unroll
        for (int t = 0; t < 8; ++t)
            C[(size_t)(row0 + t + 8 * half) * Ncols + col0 + c * 16 + cn] = acc[c][t];
}

// ---------------------------------------------------------------------------
// Post-processing: 1 wave per (n, h).
//  - mix = sigmoid(x . vmix_w[:,h] + b[h])
//  - orig_v output, v-lerp with value_residual, interleaved RoPE on q/k
//  - emit per-head f16 q/k/v  ([H][N][64])
// ---------------------------------------------------------------------------
__global__ void __launch_bounds__(128)
postproc_kernel(const float* __restrict__ qkv, const _Float16* __restrict__ xh,
                const float* __restrict__ vmix_w, const float* __restrict__ vmix_b,
                const float* __restrict__ vres, float* __restrict__ out_v,
                _Float16* __restrict__ qh, _Float16* __restrict__ kh,
                _Float16* __restrict__ vh) {
    int task = blockIdx.x * 4 + (threadIdx.x >> 5);
    if (task >= NN * HH) return;
    const int lane = threadIdx.x & 31;
    const int h = task & 31;
    const int n = task >> 5;

    // learned value-mix gate
    float acc = 0.f;
    const _Float16* xr = xh + (size_t)n * DD;
    for (int d = lane; d < DD; d += 32)
        acc += (float)xr[d] * vmix_w[(size_t)d * HH + h];
    for (int m = 16; m; m >>= 1) acc += __shfl_xor(acc, m, 32);
    const float mix = 1.f / (1.f + __expf(-(acc + vmix_b[h])));

    const int d0 = 2 * lane;                       // interleaved rope pair
    const size_t qbase = (size_t)n * M3 + h * DHD;
    float q0 = qkv[qbase + d0],            q1 = qkv[qbase + d0 + 1];
    float k0 = qkv[qbase + DD + d0],       k1 = qkv[qbase + DD + d0 + 1];
    float v0 = qkv[qbase + 2 * DD + d0],   v1 = qkv[qbase + 2 * DD + d0 + 1];

    const size_t hv = ((size_t)h * NN + n) * DHD + d0;
    out_v[hv] = v0; out_v[hv + 1] = v1;            // orig value residual output

    float vr0 = vres[hv], vr1 = vres[hv + 1];
    float vm0 = v0 + (vr0 - v0) * mix;
    float vm1 = v1 + (vr1 - v1) * mix;

    float inv = powf(10000.f, -((float)d0) / 64.f);
    float ang = (float)n * inv;
    float cs = cosf(ang), sn = sinf(ang);
    float qo0 = q0 * cs - q1 * sn, qo1 = q1 * cs + q0 * sn;
    float ko0 = k0 * cs - k1 * sn, ko1 = k1 * cs + k0 * sn;

    qh[hv] = (_Float16)qo0; qh[hv + 1] = (_Float16)qo1;
    kh[hv] = (_Float16)ko0; kh[hv + 1] = (_Float16)ko1;
    vh[hv] = (_Float16)vm0; vh[hv + 1] = (_Float16)vm1;
}

// ---------------------------------------------------------------------------
// Flash attention: 1 wave per (window, head, 16-query tile). Keys processed in
// chunks of 32 (S via 2x2 WMMAs, P.V via 4 WMMAs). P crosses C->A layout via
// wave-private LDS.
// ---------------------------------------------------------------------------
__global__ void __launch_bounds__(128)
attn_kernel(const _Float16* __restrict__ qh, const _Float16* __restrict__ kh,
            const _Float16* __restrict__ vh, const _Float16* __restrict__ pk,
            const _Float16* __restrict__ pv, _Float16* __restrict__ oh) {
    __shared__ __align__(32) _Float16 plds[4][16 * 32];
    const int wave = threadIdx.x >> 5;
    const int lane = threadIdx.x & 31;
    const int QT = 33;                          // ceil(516/16)
    int task = blockIdx.x * 4 + wave;
    if (task >= WWN * HH * QT) return;
    const int qt = task % QT;
    const int rem = task / QT;
    const int h = rem % HH;
    const int w = rem / HH;
    const int i0 = qt * 16;

    const _Float16* qb = qh + (size_t)h * NN * DHD;
    const _Float16* kb = kh + (size_t)h * NN * DHD;
    const _Float16* vb = vh + (size_t)h * NN * DHD;
    const _Float16* pkh = pk + h * PPM * DHD;
    const _Float16* pvh = pv + h * PPM * DHD;

    const int khalf = (lane >> 4) << 3;
    const int half = lane >> 4;
    const int cn = lane & 15;

    int qrow = w * SLW + i0 + cn;
    if (qrow > NN - 1) qrow = NN - 1;
    const _Float16* qp = qb + (size_t)qrow * DHD;
    v16h qf0 = frag16(qp + khalf);
    v16h qf1 = frag16(qp + 32 + khalf);

    v8f o[4] = {};
    float mrun[8], lrun[8];
#pragma unroll
    for (int r = 0; r < 8; ++r) { mrun[r] = -1e30f; lrun[r] = 0.f; }

    const int nk = (w - 1) * SLW;               // key j>=P  ->  n = nk + j - P
    const float sc = 0.125f * 1.44269504089f;   // DH^-0.5 * log2(e)

    for (int j0 = 0; j0 < 1056; j0 += 32) {
        // chunk-level mask skip
        bool hasP = (j0 < PPM);
        bool prevV = false, curV = false;
        if (w > 0 && j0 < PPM + SLW && j0 + 32 > PPM) {
            int hi = (j0 + 32 < PPM + SLW) ? j0 + 32 : PPM + SLW;
            if (hi - 1 - PPM >= i0) prevV = true;
        }
        if (j0 + 32 > PPM + SLW && j0 < NKEY) {
            int lo2 = (j0 > PPM + SLW) ? j0 : PPM + SLW;
            int imax = (i0 + 15 < SLW - 1) ? i0 + 15 : SLW - 1;
            if (lo2 - (PPM + SLW) <= imax) curV = true;
        }
        if (!(hasP || prevV || curV)) continue;

        // ---- S = Q.K^T : two 16x16 subtiles ----
        v8f sacc[2];
#pragma unroll
        for (int s = 0; s < 2; ++s) {
            int j = j0 + s * 16 + cn;
            const _Float16* kp;
            if (j < PPM) kp = pkh + j * DHD;
            else {
                int nn = nk + j - PPM;
                if (nn < 0) nn = 0;
                if (nn > NN - 1) nn = NN - 1;
                kp = kb + (size_t)nn * DHD;
            }
            v16h kf0 = frag16(kp + khalf);
            v16h kf1 = frag16(kp + 32 + khalf);
            v8f a = {};
            a = __builtin_amdgcn_wmma_f32_16x16x32_f16(false, qf0, false, kf0,
                                                       (short)0, a, false, false);
            a = __builtin_amdgcn_wmma_f32_16x16x32_f16(false, qf1, false, kf1,
                                                       (short)0, a, false, false);
            sacc[s] = a;
        }

        // ---- mask + online softmax (per row r) ----
#pragma unroll
        for (int r = 0; r < 8; ++r) {
            const int i = i0 + r + 8 * half;
#pragma unroll
            for (int s = 0; s < 2; ++s) {
                int j = j0 + s * 16 + cn;
                bool valid;
                if (j < PPM)            valid = true;
                else if (j < PPM + SLW) valid = (w > 0) && ((j - PPM) >= i);
                else if (j < NKEY)      valid = (j - PPM - SLW) <= i;
                else                    valid = false;
                sacc[s][r] = valid ? sacc[s][r] * sc : -1e30f;
            }
            float mx = fmaxf(sacc[0][r], sacc[1][r]);
            for (int m = 1; m < 16; m <<= 1) mx = fmaxf(mx, __shfl_xor(mx, m, 32));
            float mnew = fmaxf(mrun[r], mx);
            float alpha = exp2f(mrun[r] - mnew);
            float p0 = (sacc[0][r] <= -1e29f) ? 0.f : exp2f(sacc[0][r] - mnew);
            float p1 = (sacc[1][r] <= -1e29f) ? 0.f : exp2f(sacc[1][r] - mnew);
            float rs = p0 + p1;
            for (int m = 1; m < 16; m <<= 1) rs += __shfl_xor(rs, m, 32);
            lrun[r] = lrun[r] * alpha + rs;
            mrun[r] = mnew;
#pragma unroll
            for (int c = 0; c < 4; ++c) o[c][r] *= alpha;
            const int mrow = r + 8 * half;
            plds[wave][mrow * 32 + cn]      = (_Float16)p0;
            plds[wave][mrow * 32 + 16 + cn] = (_Float16)p1;
        }

        // ---- O += P.V ----  (P reloaded from LDS in A-fragment layout)
        v16h pf = frag16(&plds[wave][cn * 32 + khalf]);
#pragma unroll
        for (int c = 0; c < 4; ++c) {
            const int dh = c * 16 + cn;
            v16h vf;
#pragma unroll
            for (int t = 0; t < 16; ++t) {
                int kk = wmma_k_of_t(lane, t);
                int j = j0 + kk;
                const _Float16* vp;
                if (j < PPM) vp = pvh + j * DHD + dh;
                else {
                    int nn = nk + j - PPM;
                    if (nn < 0) nn = 0;
                    if (nn > NN - 1) nn = NN - 1;
                    vp = vb + (size_t)nn * DHD + dh;
                }
                vf[t] = *vp;
            }
            o[c] = __builtin_amdgcn_wmma_f32_16x16x32_f16(false, pf, false, vf,
                                                          (short)0, o[c], false, false);
        }
    }

    // ---- finalize ----
#pragma unroll
    for (int r = 0; r < 8; ++r) {
        const int i = i0 + r + 8 * half;
        if (i >= SLW) continue;
        float linv = (lrun[r] > 0.f) ? 1.f / lrun[r] : 0.f;
        const int n = w * SLW + i;
#pragma unroll
        for (int c = 0; c < 4; ++c)
            oh[(size_t)n * DD + h * DHD + c * 16 + cn] = (_Float16)(o[c][r] * linv);
    }
}

// ---------------------------------------------------------------------------
extern "C" void kernel_launch(void* const* d_in, const int* in_sizes, int n_in,
                              void* d_out, int out_size, void* d_ws, size_t ws_size,
                              hipStream_t stream) {
    const float* hs     = (const float*)d_in[0];  // [1,N,D]
    const float* vres   = (const float*)d_in[1];  // [1,H,N,DH]
    const float* norm_w = (const float*)d_in[2];  // [D]
    const float* w_qkv  = (const float*)d_in[3];  // [D, 3*H*DH]
    const float* w_out  = (const float*)d_in[4];  // [H*DH, D]
    const float* vmix_w = (const float*)d_in[5];  // [D, H]
    const float* vmix_b = (const float*)d_in[6];  // [H]
    const float* pmem   = (const float*)d_in[7];  // [2,H,P,DH]
    float* out  = (float*)d_out;                  // [N*D] ++ [H*N*DH]
    float* outv = out + (size_t)NN * DD;

    // workspace carve-out (256B-aligned)
    char* ws = (char*)d_ws;
    size_t off = 0;
    auto carve = [&](size_t bytes) {
        char* p = ws + off;
        off = (off + bytes + 255) & ~(size_t)255;
        return p;
    };
    _Float16* xh    = (_Float16*)carve((size_t)NN * DD * 2);
    _Float16* wqkvp = (_Float16*)carve((size_t)DD * M3 * 2);
    _Float16* woutp = (_Float16*)carve((size_t)DD * DD * 2);
    float*    qkv   = (float*)   carve((size_t)NN * M3 * 4);
    _Float16* qhb   = (_Float16*)carve((size_t)HH * NN * DHD * 2);
    _Float16* khb   = (_Float16*)carve((size_t)HH * NN * DHD * 2);
    _Float16* vhb   = (_Float16*)carve((size_t)HH * NN * DHD * 2);
    _Float16* ohb   = (_Float16*)carve((size_t)NN * DD * 2);
    _Float16* pkb   = (_Float16*)carve((size_t)HH * PPM * DHD * 2);
    _Float16* pvb   = (_Float16*)carve((size_t)HH * PPM * DHD * 2);
    (void)ws_size; (void)n_in; (void)in_sizes; (void)out_size;

    // 1) RMSNorm -> f16 x
    rmsnorm_kernel<<<NN, 256, 0, stream>>>(hs, norm_w, xh);

    // 2) pack weights into WMMA B-fragment layout
    {
        size_t tot = (size_t)DD * M3;
        pack_b_kernel<<<(unsigned)((tot + 255) / 256), 256, 0, stream>>>(w_qkv, wqkvp, DD, M3);
        size_t tot2 = (size_t)DD * DD;
        pack_b_kernel<<<(unsigned)((tot2 + 255) / 256), 256, 0, stream>>>(w_out, woutp, DD, DD);
        pm_kernel<<<(HH * PPM * DHD + 255) / 256, 256, 0, stream>>>(pmem, pkb, pvb);
    }

    // 3) QKV projection: [N,D] x [D,6144]
    {
        dim3 grid((NN / 16 + 3) / 4, M3 / 64);
        wmma_gemm_kernel<<<grid, 128, 0, stream>>>(xh, wqkvp, qkv, NN, M3, DD);
    }

    // 4) v-mix gate + orig_v output + RoPE + per-head f16 q/k/v
    {
        int tasks = NN * HH;
        postproc_kernel<<<(tasks + 3) / 4, 128, 0, stream>>>(
            qkv, xh, vmix_w, vmix_b, vres, outv, qhb, khb, vhb);
    }

    // 5) sliding-window flash attention with persistent memory
    {
        int tasks = WWN * HH * 33;
        attn_kernel<<<(tasks + 3) / 4, 128, 0, stream>>>(qhb, khb, vhb, pkb, pvb, ohb);
    }

    // 6) output projection: [N,2048] x [2048,2048] -> d_out
    {
        dim3 grid((NN / 16 + 3) / 4, DD / 64);
        wmma_gemm_kernel<<<grid, 128, 0, stream>>>(ohb, woutp, out, NN, DD, DD);
    }
}
